// MultiHeadAttentionLayer_23708219474775
// MI455X (gfx1250) — matrix operand order repaired and checked by reading, not compiled, not run on hardware
//
#include <hip/hip_runtime.h>
#include <math.h>
#include <stdint.h>

// MHA layer for MI455X (gfx1250, wave32).
// All matmuls via v_wmma_f32_16x16x32_f16; shared tiles staged into LDS with
// quad-buffered global_load_async_to_lds_b128 (ASYNCcnt, prefetch distance 2,
// branchless staging + uniform partial s_wait_asynccnt) and consumed via
// batched ds_load_b128; A fragments are register-pipelined one K-step ahead.
//
// Workspace layout (bytes):
//   [0,8M)    Xh   : f16 X, [4096][1024]
//   [8M,10M)  WqT  : f16 Wq^T [1024 out][1024 in]
//   [10M,12M) WkT
//   [12M,14M) WvT
//   [14M,16M) WpT  : Wpost^T
//   [16M,24M) Qh   : f16 [b][h][2048][64]
//   [24M,32M) Kh   : f16 [b][h][2048][64]  (pre-scaled by 1/sqrt(64))
//   [32M,40M) Vt   : f16 [b][h][64][2048]  (transposed per head)
//   [40M,48M) Ah   : f16 gelu(attn out) [4096][1024]

typedef _Float16 f16;
typedef __attribute__((ext_vector_type(16))) _Float16 v16h;
typedef __attribute__((ext_vector_type(8)))  _Float16 v8h;
typedef __attribute__((ext_vector_type(8)))  float    v8f;

#define WMMA(a,b,c) __builtin_amdgcn_wmma_f32_16x16x32_f16(false,(a),false,(b),(short)0,(c),false,false)

static __device__ __forceinline__ v16h frag16(const f16* p0, const f16* p1) {
  v8h lo = *(const v8h*)p0;
  v8h hi = *(const v8h*)p1;
  v16h r;
#pragma unroll
  for (int i = 0; i < 8; ++i) { r[i] = lo[i]; r[8 + i] = hi[i]; }
  return r;
}

// 16-byte async copy global -> LDS (tracked by ASYNCcnt, no VGPR data path).
// No "memory" clobber: issue/wait ordering is asm-volatile vs asm-volatile, and
// LDS consumer ordering is provided by __syncthreads()'s workgroup fence.
static __device__ __forceinline__ void async_b128(f16* lds_dst, const f16* gsrc) {
  uint32_t loff = (uint32_t)(uintptr_t)lds_dst;   // low 32 bits = LDS byte offset
  asm volatile("global_load_async_to_lds_b128 %0, %1, off"
               :: "v"(loff), "v"((uint64_t)(uintptr_t)gsrc));
}
template <int N>
static __device__ __forceinline__ void wait_async() {
  asm volatile("s_wait_asynccnt %0" :: "n"(N));
}

// ---------------- conversion kernels ----------------

__global__ __launch_bounds__(256) void cvt_x(const float* __restrict__ X,
                                             f16* __restrict__ Xh, int n) {
  int i = blockIdx.x * 256 + threadIdx.x;
  if (i < n) Xh[i] = (f16)X[i];
}

__global__ __launch_bounds__(256) void cvt_wt(const float* __restrict__ Wq,
                                              const float* __restrict__ Wk,
                                              const float* __restrict__ Wv,
                                              const float* __restrict__ Wp,
                                              f16* __restrict__ WqT,
                                              f16* __restrict__ WkT,
                                              f16* __restrict__ WvT,
                                              f16* __restrict__ WpT) {
  int idx = blockIdx.x * 256 + threadIdx.x;       // 4 * 1024 * 1024 total
  int m = idx >> 20;
  int r = idx & ((1 << 20) - 1);
  int o = r >> 10;      // out col (row of W^T)
  int k = r & 1023;     // in row
  const float* src = (m == 0) ? Wq : (m == 1) ? Wk : (m == 2) ? Wv : Wp;
  f16* dst = (m == 0) ? WqT : (m == 1) ? WkT : (m == 2) ? WvT : WpT;
  dst[o * 1024 + k] = (f16)src[k * 1024 + o];
}

// ---------------- fused QKV projection ----------------
// Block = 8 waves: 128 rows x 64 cols. Grid: 3 mats * 32 row-chunks * 16 ngroups.
__global__ __launch_bounds__(256) void qkv_gemm(const f16* __restrict__ Xh,
                                                const f16* __restrict__ WqT,
                                                const f16* __restrict__ WkT,
                                                const f16* __restrict__ WvT,
                                                f16* __restrict__ Qh,
                                                f16* __restrict__ Kh,
                                                f16* __restrict__ Vt) {
  __shared__ f16 bt[4][64 * 32];                  // quad-buffered B tile, 16KB
  int tid = threadIdx.x;
  int wib = tid >> 5, lane = tid & 31, n16 = lane & 15, h2 = lane >> 4;

  int bx  = blockIdx.x;
  int mat = bx / 512;              // 0=Q 1=K 2=V
  int rem = bx - mat * 512;
  int mc  = rem >> 4;              // 0..31 (128-row chunk)
  int ng  = rem & 15;              // 0..15 (64-col group)

  const f16* WT = (mat == 0) ? WqT : (mat == 1) ? WkT : WvT;
  const f16* wsrc = WT + (size_t)(ng * 64 + (tid >> 2)) * 1024 + (tid & 3) * 8;
  f16* sdst = &bt[0][(tid >> 2) * 32 + (tid & 3) * 8];

  int rowbase = mc * 128 + wib * 16;
  const f16* xrow = Xh + (size_t)(rowbase + n16) * 1024;

  v8f c[4];
#pragma unroll
  for (int nt = 0; nt < 4; ++nt) c[nt] = (v8f){};

  async_b128(sdst,        wsrc);                  // stage k0=0  -> buf 0
  async_b128(sdst + 2048, wsrc + 32);             // stage k0=32 -> buf 1
  v16h a = frag16(xrow + 16 * h2, xrow + 16 * h2 + 8);   // A for k0=0

  for (int k0 = 0; k0 < 1024; k0 += 32) {
    int it  = k0 >> 5;
    int buf = it & 3;
    wait_async<1>();     // oldest outstanding stage == current tile: resident
    __syncthreads();
    // branchless prefetch distance 2 (tail clamps source; dead buffers)
    int kp = k0 + 64 < 1024 ? k0 + 64 : 960;
    async_b128(sdst + ((it + 2) & 3) * 2048, wsrc + kp);

    // batch all B-fragment LDS loads, then pipeline next A from global
    const f16* bp = &bt[buf][0] + n16 * 32 + 16 * h2;
    v16h b0 = frag16(bp,           bp + 8);
    v16h b1 = frag16(bp + 16 * 32, bp + 16 * 32 + 8);
    v16h b2 = frag16(bp + 32 * 32, bp + 32 * 32 + 8);
    v16h b3 = frag16(bp + 48 * 32, bp + 48 * 32 + 8);
    int ka = k0 + 32 < 1024 ? k0 + 32 : 0;        // clamped (redundant on tail)
    v16h a_next = frag16(xrow + ka + 16 * h2, xrow + ka + 16 * h2 + 8);

    c[0] = WMMA(a, b0, c[0]);
    c[1] = WMMA(a, b1, c[1]);
    c[2] = WMMA(a, b2, c[2]);
    c[3] = WMMA(a, b3, c[3]);
    a = a_next;
  }

  int bb = rowbase >> 11;          // batch
  int sl = rowbase & 2047;         // local seq base
#pragma unroll
  for (int nt = 0; nt < 4; ++nt) {
    int col = ng * 64 + nt * 16 + n16;
    int hh = col >> 6, d = col & 63;
    if (mat < 2) {
      float ksc = (mat == 1) ? 0.125f : 1.0f;   // 1/sqrt(64) folded into K
      f16* dst = ((mat == 0) ? Qh : Kh) +
                 ((size_t)(bb * 16 + hh) * 2048 + sl + 8 * h2) * 64 + d;
#pragma unroll
      for (int r = 0; r < 8; ++r) dst[r * 64] = (f16)(c[nt][r] * ksc);
    } else {
      v8h pk;
#pragma unroll
      for (int r = 0; r < 8; ++r) pk[r] = (f16)c[nt][r];
      *(v8h*)(Vt + ((size_t)(bb * 16 + hh) * 64 + d) * 2048 + sl + 8 * h2) = pk;
    }
  }
}

// ---------------- flash attention + GELU ----------------
// 512 blocks; the 8 waves of a block share (batch, head) and stream the same
// K/V tiles, staged once per block into quad-buffered LDS.
__global__ __launch_bounds__(256) void attn(const f16* __restrict__ Qh,
                                            const f16* __restrict__ Kh,
                                            const f16* __restrict__ Vt,
                                            f16* __restrict__ Ah) {
  __shared__ f16 kt[4][32 * 64];                  // K tile  [key][dim], 16KB
  __shared__ f16 vtl[4][64 * 32];                 // V tile  [dim][key], 16KB
  __shared__ f16 pb[8][16 * 32];                  // per-wave P tile, 8KB
  int tid = threadIdx.x;
  int wib = tid >> 5, lane = tid & 31, n16 = lane & 15, h2 = lane >> 4;

  int bx = blockIdx.x;
  int b  = bx >> 8;
  int h  = (bx >> 4) & 15;
  int qt = (bx & 15) * 8 + wib;

  const f16* Qp = Qh + ((size_t)(b * 16 + h) * 2048 + qt * 16) * 64;
  const f16* Kp = Kh + (size_t)(b * 16 + h) * 2048 * 64;
  const f16* Vp = Vt + (size_t)(b * 16 + h) * 64 * 2048;

  const f16* ksrc = Kp + (size_t)(tid >> 3) * 64 + (tid & 7) * 8;   // + t0*64
  f16* kdst = &kt[0][(tid >> 3) * 64 + (tid & 7) * 8];
  const f16* vsrc = Vp + (size_t)(tid >> 2) * 2048 + (tid & 3) * 8; // + t0
  f16* vdst = &vtl[0][(tid >> 2) * 32 + (tid & 3) * 8];
  f16* pbuf = &pb[wib][0];

  const f16* qr = Qp + n16 * 64;
  v16h aq0 = frag16(qr + 16 * h2,      qr + 16 * h2 + 8);   // K = 0..31
  v16h aq1 = frag16(qr + 32 + 16 * h2, qr + 40 + 16 * h2);   // K = 32..63

  float mi[8], li[8];
  v8f acc[4];
#pragma unroll
  for (int r = 0; r < 8; ++r) { mi[r] = -3.0e38f; li[r] = 0.0f; }
#pragma unroll
  for (int nt = 0; nt < 4; ++nt) acc[nt] = (v8f){};

  async_b128(kdst,        ksrc);                  // t0=0  -> buf 0
  async_b128(vdst,        vsrc);
  async_b128(kdst + 2048, ksrc + 32 * 64);        // t0=32 -> buf 1
  async_b128(vdst + 2048, vsrc + 32);
  for (int t0 = 0; t0 < 2048; t0 += 32) {
    int it  = t0 >> 5;
    int buf = it & 3;
    wait_async<2>();     // oldest two (current K+V) resident
    __syncthreads();
    // branchless prefetch distance 2 (tail clamps source; dead buffers)
    int tp = t0 + 64 < 2048 ? t0 + 64 : 1984;
    int nb = (it + 2) & 3;
    async_b128(kdst + nb * 2048, ksrc + (size_t)tp * 64);
    async_b128(vdst + nb * 2048, vsrc + tp);

    // batch all K-fragment LDS loads, then the 4 S-WMMAs back-to-back
    const f16* kb  = &kt[buf][0];
    const f16* k0p = kb + n16 * 64 + 16 * h2;
    const f16* k1p = kb + (16 + n16) * 64 + 16 * h2;
    v16h bk00 = frag16(k0p,      k0p + 8);
    v16h bk01 = frag16(k0p + 32, k0p + 40);
    v16h bk10 = frag16(k1p,      k1p + 8);
    v16h bk11 = frag16(k1p + 32, k1p + 40);
    v8f s0 = (v8f){}, s1 = (v8f){};
    s0 = WMMA(aq0, bk00, s0);
    s1 = WMMA(aq0, bk10, s1);
    s0 = WMMA(aq1, bk01, s0);
    s1 = WMMA(aq1, bk11, s1);

    // online softmax per row (row r+8*h2 lives across the 16-lane group)
    float sc[8];
#pragma unroll
    for (int r = 0; r < 8; ++r) {
      float mx = fmaxf(s0[r], s1[r]);
      mx = fmaxf(mx, __shfl_xor(mx, 1));
      mx = fmaxf(mx, __shfl_xor(mx, 2));
      mx = fmaxf(mx, __shfl_xor(mx, 4));
      mx = fmaxf(mx, __shfl_xor(mx, 8));
      float mn = fmaxf(mi[r], mx);
      sc[r] = __expf(mi[r] - mn);
      float p0 = __expf(s0[r] - mn);
      float p1 = __expf(s1[r] - mn);
      float ls = p0 + p1;
      ls += __shfl_xor(ls, 1);
      ls += __shfl_xor(ls, 2);
      ls += __shfl_xor(ls, 4);
      ls += __shfl_xor(ls, 8);
      li[r] = li[r] * sc[r] + ls;
      mi[r] = mn;
      pbuf[(r + 8 * h2) * 32 + n16]      = (f16)p0;   // P tile, C->LDS
      pbuf[(r + 8 * h2) * 32 + 16 + n16] = (f16)p1;
    }

    // batch V-fragment loads (independent of P stores; DS is in-order so the
    // P reload below still sees the stores), rescale acc meanwhile
    const f16* vp = &vtl[buf][0] + n16 * 32 + 16 * h2;
    v16h bv0 = frag16(vp,           vp + 8);
    v16h bv1 = frag16(vp + 16 * 32, vp + 16 * 32 + 8);
    v16h bv2 = frag16(vp + 32 * 32, vp + 32 * 32 + 8);
    v16h bv3 = frag16(vp + 48 * 32, vp + 48 * 32 + 8);

#pragma unroll
    for (int nt = 0; nt < 4; ++nt)
#pragma unroll
      for (int r = 0; r < 8; ++r) acc[nt][r] *= sc[r];

    // reload P as A-fragment (16x32), then O += P*V
    const f16* pr = pbuf + n16 * 32;
    v16h ap = frag16(pr + 16 * h2, pr + 16 * h2 + 8);
    acc[0] = WMMA(ap, bv0, acc[0]);
    acc[1] = WMMA(ap, bv1, acc[1]);
    acc[2] = WMMA(ap, bv2, acc[2]);
    acc[3] = WMMA(ap, bv3, acc[3]);
  }

  // normalize, exact-erf GELU, store f16 A [4096][1024]
#pragma unroll
  for (int nt = 0; nt < 4; ++nt)
#pragma unroll
    for (int r = 0; r < 8; ++r) {
      float o = acc[nt][r] / li[r];
      float g = 0.5f * o * (1.0f + erff(o * 0.70710678118654752f));
      int s = qt * 16 + r + 8 * h2;
      Ah[(size_t)(b * 2048 + s) * 1024 + h * 64 + nt * 16 + n16] = (f16)g;
    }
}

// ---------------- post projection + residual ----------------
// Block = 8 waves: 128 rows x 64 cols. Grid: 32 row-chunks * 16 ngroups = 512.
__global__ __launch_bounds__(256) void post_gemm(const f16* __restrict__ Ah,
                                                 const f16* __restrict__ WpT,
                                                 const float* __restrict__ X,
                                                 float* __restrict__ Out) {
  __shared__ f16 bt[4][64 * 32];
  int tid = threadIdx.x;
  int wib = tid >> 5, lane = tid & 31, n16 = lane & 15, h2 = lane >> 4;

  int bx = blockIdx.x;
  int mc = bx >> 4;
  int ng = bx & 15;

  const f16* wsrc = WpT + (size_t)(ng * 64 + (tid >> 2)) * 1024 + (tid & 3) * 8;
  f16* sdst = &bt[0][(tid >> 2) * 32 + (tid & 3) * 8];

  int rowbase = mc * 128 + wib * 16;
  const f16* arow = Ah + (size_t)(rowbase + n16) * 1024;

  v8f c[4];
#pragma unroll
  for (int nt = 0; nt < 4; ++nt) c[nt] = (v8f){};

  async_b128(sdst,        wsrc);
  async_b128(sdst + 2048, wsrc + 32);
  v16h a = frag16(arow + 16 * h2, arow + 16 * h2 + 8);

  for (int k0 = 0; k0 < 1024; k0 += 32) {
    int it  = k0 >> 5;
    int buf = it & 3;
    wait_async<1>();
    __syncthreads();
    int kp = k0 + 64 < 1024 ? k0 + 64 : 960;
    async_b128(sdst + ((it + 2) & 3) * 2048, wsrc + kp);

    const f16* bp = &bt[buf][0] + n16 * 32 + 16 * h2;
    v16h b0 = frag16(bp,           bp + 8);
    v16h b1 = frag16(bp + 16 * 32, bp + 16 * 32 + 8);
    v16h b2 = frag16(bp + 32 * 32, bp + 32 * 32 + 8);
    v16h b3 = frag16(bp + 48 * 32, bp + 48 * 32 + 8);
    int ka = k0 + 32 < 1024 ? k0 + 32 : 0;
    v16h a_next = frag16(arow + ka + 16 * h2, arow + ka + 16 * h2 + 8);

    c[0] = WMMA(a, b0, c[0]);
    c[1] = WMMA(a, b1, c[1]);
    c[2] = WMMA(a, b2, c[2]);
    c[3] = WMMA(a, b3, c[3]);
    a = a_next;
  }

#pragma unroll
  for (int nt = 0; nt < 4; ++nt) {
    int col = ng * 64 + nt * 16 + n16;
#pragma unroll
    for (int r = 0; r < 8; ++r) {
      size_t row = (size_t)rowbase + r + 8 * h2;
      Out[row * 1024 + col] = X[row * 1024 + col] + c[nt][r];
    }
  }
}

// ---------------- launch ----------------

extern "C" void kernel_launch(void* const* d_in, const int* in_sizes, int n_in,
                              void* d_out, int out_size, void* d_ws, size_t ws_size,
                              hipStream_t stream) {
  const float* X  = (const float*)d_in[0];
  const float* Wq = (const float*)d_in[1];
  const float* Wk = (const float*)d_in[2];
  const float* Wv = (const float*)d_in[3];
  const float* Wp = (const float*)d_in[4];

  char* ws = (char*)d_ws;
  const size_t MB = 1u << 20;
  f16* Xh  = (f16*)(ws + 0 * MB);
  f16* WqT = (f16*)(ws + 8 * MB);
  f16* WkT = (f16*)(ws + 10 * MB);
  f16* WvT = (f16*)(ws + 12 * MB);
  f16* WpT = (f16*)(ws + 14 * MB);
  f16* Qh  = (f16*)(ws + 16 * MB);
  f16* Kh  = (f16*)(ws + 24 * MB);
  f16* Vt  = (f16*)(ws + 32 * MB);
  f16* Ah  = (f16*)(ws + 40 * MB);

  cvt_x<<<(4096 * 1024 + 255) / 256, 256, 0, stream>>>(X, Xh, 4096 * 1024);
  cvt_wt<<<(4 * 1024 * 1024) / 256, 256, 0, stream>>>(Wq, Wk, Wv, Wp,
                                                      WqT, WkT, WvT, WpT);
  qkv_gemm<<<1536, 256, 0, stream>>>(Xh, WqT, WkT, WvT, Qh, Kh, Vt);
  attn<<<512, 256, 0, stream>>>(Qh, Kh, Vt, Ah);
  post_gemm<<<512, 256, 0, stream>>>(Ah, WpT, X, (float*)d_out);
}
